// PivotGraphLearner_45174466019847
// MI455X (gfx1250) — matrix operand order repaired and checked by reading, not compiled, not run on hardware
//
#include <hip/hip_runtime.h>

typedef __attribute__((ext_vector_type(16))) __bf16 v16bf;
typedef __attribute__((ext_vector_type(8)))  float  v8f;

union ABfrag { v16bf v; uint4 q[2]; };

#define N_NODES 50000
#define M_PIV   2048
#define D_DIM   256
#define P_PROJ  8
#define K_TOT   2048            // P*D
#define TOPK    32
#define ROWS_WG 32
#define A_STRIDE_B 4112         // 2048 bf16 * 2 + 16B pad (bank-conflict-free row gather)
#define LDS_BYTES 262144        // 32*2048*4 (attention tile) >= A(131584)+stageN(32768)+stageW(8192)

__device__ __forceinline__ unsigned short f2bf(float f) {
  union { float f; unsigned u; } c; c.f = f;
  unsigned u = c.u + 0x7FFFu + ((c.u >> 16) & 1u);   // RNE
  return (unsigned short)(u >> 16);
}
__device__ __forceinline__ unsigned f2ord(float f) {   // order-preserving float->uint
  union { float f; unsigned u; } c; c.f = f;
  return (c.u & 0x80000000u) ? ~c.u : (c.u | 0x80000000u);
}
__device__ __forceinline__ float ord2f(unsigned k) {
  union { unsigned u; float f; } c;
  c.u = (k & 0x80000000u) ? (k & 0x7FFFFFFFu) : ~k;
  return c.f;
}

// ---------------- Kernel 1: normalize pivots -> Y (bf16, M x K, 1/P folded) ----
__global__ __launch_bounds__(256) void pivot_norm_kernel(
    const float* __restrict__ pivots, const float* __restrict__ W,
    unsigned short* __restrict__ Y) {
  __shared__ float sp[D_DIM];
  __shared__ float sw[P_PROJ * D_DIM];
  __shared__ float red[8];
  __shared__ float bc;
  const int t = threadIdx.x;
  const int m = blockIdx.x;
  sp[t] = pivots[(size_t)m * D_DIM + t];
  for (int i = t; i < P_PROJ * D_DIM; i += 256) sw[i] = W[i];
  __syncthreads();
  for (int p = 0; p < P_PROJ; ++p) {
    float x = sw[p * D_DIM + t] * sp[t];
    float s = x * x;
    for (int msk = 16; msk; msk >>= 1) s += __shfl_xor(s, msk, 32);
    if ((t & 31) == 0) red[t >> 5] = s;
    __syncthreads();
    if (t == 0) { float tot = 0.f; for (int i = 0; i < 8; ++i) tot += red[i]; bc = tot; }
    __syncthreads();
    float sc = 1.0f / (fmaxf(sqrtf(bc), 1e-12f) * (float)P_PROJ);
    Y[(size_t)m * K_TOT + p * D_DIM + t] = f2bf(x * sc);
    __syncthreads();
  }
}

// -------- Kernel 2: fused normalize(A) -> bf16 WMMA GEMM -> exact top-32 mask --
__global__ __launch_bounds__(512) void pivot_attn_topk_kernel(
    const float* __restrict__ nodes, const float* __restrict__ W,
    const unsigned short* __restrict__ Y, float* __restrict__ out) {
  extern __shared__ char smem[];
  const int tid  = threadIdx.x;
  const int lane = tid & 31;
  const int wv   = tid >> 5;          // 0..15
  const int n0   = blockIdx.x * ROWS_WG;

  float* stN = (float*)(smem + ROWS_WG * A_STRIDE_B);            // 32x256 f32
  float* stW = (float*)(smem + ROWS_WG * A_STRIDE_B + 32768);    // 8x256 f32

  // phase 0: stage node rows (clamped) + W
  for (int i = tid; i < ROWS_WG * D_DIM; i += 512) {
    int r = i >> 8, d = i & 255;
    int n = n0 + r; if (n > N_NODES - 1) n = N_NODES - 1;
    stN[i] = nodes[(size_t)n * D_DIM + d];
  }
  for (int i = tid; i < P_PROJ * D_DIM; i += 512) stW[i] = W[i];
  __syncthreads();

  // phase 1: per (row,p) normalize -> bf16 A tile in LDS (row stride 4112 B)
  if (tid < ROWS_WG * P_PROJ) {
    int r = tid >> 3, p = tid & 7;
    const float* xn = stN + r * D_DIM;
    const float* wp = stW + p * D_DIM;
    float ss = 0.f;
    for (int d = 0; d < D_DIM; ++d) { float x = wp[d] * xn[d]; ss += x * x; }
    float sc = 1.0f / fmaxf(sqrtf(ss), 1e-12f);
    char* arow = smem + r * A_STRIDE_B + p * (D_DIM * 2);
    for (int d = 0; d < D_DIM; d += 4) {
      unsigned short h0 = f2bf(wp[d+0]*xn[d+0]*sc), h1 = f2bf(wp[d+1]*xn[d+1]*sc);
      unsigned short h2 = f2bf(wp[d+2]*xn[d+2]*sc), h3 = f2bf(wp[d+3]*xn[d+3]*sc);
      uint2 pk; pk.x = (unsigned)h0 | ((unsigned)h1 << 16);
                pk.y = (unsigned)h2 | ((unsigned)h3 << 16);
      *(uint2*)(arow + d * 2) = pk;
    }
  }
  __syncthreads();

  // phase 2: GEMM. wave -> 16 rows x 256 cols = 16 WMMA tiles, K = 2048.
  // B fragments are software-pipelined 3 tiles ahead through a 4-deep
  // rotating register buffer so global_load_b128s overlap the WMMAs.
  const int rowbase = (wv >> 3) * 16;
  const int colbase = (wv & 7) * 256;
  const int hi = lane >> 4, lo = lane & 15;
  const char* aP = smem + (rowbase + lo) * A_STRIDE_B + hi * 32;
  const char* yB = (const char*)Y + ((size_t)(colbase + lo) * K_TOT + hi * 16) * 2;

#define LDB(buf, t, k) {                                              \
    const char* _p = yB + (t) * (16 * K_TOT * 2) + (k) * 2;           \
    (buf).q[0] = *(const uint4*)_p;                                   \
    (buf).q[1] = *(const uint4*)(_p + 16); }

  v8f acc[16] = {};
  ABfrag bb[4];
  LDB(bb[0], 0, 0)
  LDB(bb[1], 1, 0)
  LDB(bb[2], 2, 0)

  for (int k0 = 0; k0 < K_TOT; k0 += 32) {
    ABfrag a;
    a.q[0] = *(const uint4*)(aP + k0 * 2);
    a.q[1] = *(const uint4*)(aP + k0 * 2 + 16);
    const int knext = (k0 + 32 < K_TOT) ? (k0 + 32) : 0;  // tail loads wrap (harmless)
    __builtin_prefetch(yB + knext * 2 + 64, 0, 0);
#pragma unroll
    for (int t = 0; t < 16; ++t) {
      int nt = t + 3;
      int nk = k0;
      if (nt >= 16) { nt -= 16; nk = knext; }
      LDB(bb[(t + 3) & 3], nt, nk)
      acc[t] = __builtin_amdgcn_wmma_f32_16x16x32_bf16(
          false, a.v, false, bb[t & 3].v, (short)0, acc[t], false, false);
    }
  }
  __syncthreads();

  // phase 3: spill attention tile (32 x 2048 f32) into LDS (overwrites A)
  float* att = (float*)smem;
#pragma unroll
  for (int t = 0; t < 16; ++t) {
    int col = colbase + t * 16 + lo;
    int rb  = rowbase + hi * 8;
#pragma unroll
    for (int v = 0; v < 8; ++v) att[(rb + v) * M_PIV + col] = acc[t][v];
  }
  __syncthreads();

  // phase 4: exact top-32 per row via 32-bit radix select, write masked row
  for (int rr = 0; rr < 2; ++rr) {
    int row = wv * 2 + rr;
    unsigned keys[64];
#pragma unroll
    for (int j = 0; j < 64; ++j) keys[j] = f2ord(att[row * M_PIV + lane + 32 * j]);
    unsigned pref = 0u;
    for (int bit = 31; bit >= 0; --bit) {
      unsigned cand = pref | (1u << bit);
      int c = 0;
#pragma unroll
      for (int j = 0; j < 64; ++j) c += (keys[j] >= cand) ? 1 : 0;
      for (int msk = 16; msk; msk >>= 1) c += __shfl_xor(c, msk, 32);
      if (c >= TOPK) pref = cand;
    }
    int n = n0 + row;
    if (n < N_NODES) {
      float* orow = out + (size_t)n * M_PIV;
#pragma unroll
      for (int j = 0; j < 64; ++j) {
        int col = lane + 32 * j;
        orow[col] = (keys[j] >= pref) ? ord2f(keys[j]) : 0.0f;
      }
    }
  }
}

extern "C" void kernel_launch(void* const* d_in, const int* in_sizes, int n_in,
                              void* d_out, int out_size, void* d_ws, size_t ws_size,
                              hipStream_t stream) {
  const float* nodes  = (const float*)d_in[0];
  const float* pivots = (const float*)d_in[1];
  const float* W      = (const float*)d_in[2];
  unsigned short* Y   = (unsigned short*)d_ws;   // M x K bf16 = 8.4 MB
  float* out          = (float*)d_out;

  pivot_norm_kernel<<<M_PIV, 256, 0, stream>>>(pivots, W, Y);

  const int nblocks = (N_NODES + ROWS_WG - 1) / ROWS_WG;  // 1563
  pivot_attn_topk_kernel<<<nblocks, 512, LDS_BYTES, stream>>>(nodes, W, Y, out);
}
